// SpatialAttnBlock_52140902973456
// MI455X (gfx1250) — matrix operand order repaired and checked by reading, not compiled, not run on hardware
//
#include <hip/hip_runtime.h>
#include <hip/hip_fp16.h>
#include <math.h>
#include <stdint.h>

// ---------------------------------------------------------------------------
// Problem constants (from reference): B=8, C=96, D=8, H=W=32, HW=1024,
// DIM=768, NH=4, DH=192.  Rows = B*HW = 8192.
// ---------------------------------------------------------------------------
#define BATCH 8
#define CCH   96
#define DD_   8
#define HH_   32
#define WW_   32
#define HWSZ  1024
#define DIMSZ 768
#define NHEAD 4
#define DHEAD 192
#define ROWS  (BATCH * HWSZ)   // 8192
#define QKVN  (3 * DIMSZ)      // 2304

typedef __attribute__((ext_vector_type(16))) _Float16 v16h;
typedef __attribute__((ext_vector_type(8)))  _Float16 v8h;
typedef __attribute__((ext_vector_type(8)))  float    v8f;
typedef __attribute__((ext_vector_type(4)))  unsigned int v4u;
typedef __attribute__((ext_vector_type(4)))  int      v4i;
typedef __attribute__((ext_vector_type(8)))  int      v8i;

#if __has_builtin(__builtin_amdgcn_tensor_load_to_lds) && __has_builtin(__builtin_amdgcn_s_wait_tensorcnt)
#define USE_TDM 1
#else
#define USE_TDM 0
#endif

// ---------------------------------------------------------------------------
// CDNA5 WMMA fragment helpers (wave32).  Layouts used by these helpers:
//   A (f16 16x32): lane l -> row M=l&15; elems e -> K = kb + e,
//                  kb = (l>>4)*16 (same contiguous K order as B).
//   B (f16 32x16): lane l -> col N=l&15; elems e -> K = e + (l>>4)*16.
//   C/D (f32 16x16): elem r -> M = r + (l>>4)*8, N = l&15.
// LDS tiles are laid out so every fragment load is two contiguous 16-byte
// ds_load_b128 per lane (row strides are multiples of 8 halves).
// ---------------------------------------------------------------------------
union U16 { v16h v; v8h h[2]; };

__device__ __forceinline__ v16h load_a_frag(const _Float16* T, int ldt, int r0, int k0, int lane) {
    int row = r0 + (lane & 15);
    int kb  = k0 + ((lane >> 4) << 4);
    const v8h* p = (const v8h*)(T + row * ldt + kb);
    U16 u; u.h[0] = p[0]; u.h[1] = p[1];   // K kb..kb+15 contiguous
    return u.v;
}
// B tile stored column-major in LDS: T[n][k], stride ldt halves.
__device__ __forceinline__ v16h load_b_frag(const _Float16* T, int ldt, int n0, int k0, int lane) {
    int col = n0 + (lane & 15);
    int kb  = k0 + ((lane >> 4) << 4);
    const v8h* p = (const v8h*)(T + col * ldt + kb);
    U16 u; u.h[0] = p[0]; u.h[1] = p[1];   // K kb..kb+15 contiguous
    return u.v;
}
__device__ __forceinline__ v8f wmma16(v16h a, v16h b, v8f c) {
    return __builtin_amdgcn_wmma_f32_16x16x32_f16(false, a, false, b, (short)0, c, false, false);
}

// ---------------------------------------------------------------------------
// Kernel 1: qkv GEMM.  out[n, j] = sum_k (x_t[n,k] + pos[n%HW,k]) * qkv_w[j,k]
// Block tile: 64(M) x 128(N) x 32(K); 8 waves; each wave = 1 Mtile x 4 Ntiles.
// ---------------------------------------------------------------------------
#define MB 64
#define NB 128
#define KB 32
#define LDA 40   // 32 + 8 pad (halves); rows 16B aligned
#define LDB 40

__global__ __launch_bounds__(256) void k_qkv_gemm(const float* __restrict__ x,
                                                  const float* __restrict__ pos,
                                                  const float* __restrict__ w,
                                                  _Float16* __restrict__ qb,
                                                  _Float16* __restrict__ kb,
                                                  _Float16* __restrict__ vb) {
    __shared__ _Float16 As[MB * LDA];   // [m][k]
    __shared__ _Float16 Bs[NB * LDB];   // [n][k]  (column-major tile of W^T)
    const int tid = threadIdx.x, lane = tid & 31, wv = tid >> 5;
    const int mt = wv & 3, ng = wv >> 2;
    const int m0 = blockIdx.y * MB;
    const int n0 = blockIdx.x * NB;
    v8f acc[4] = {};
    for (int k0 = 0; k0 < DIMSZ; k0 += KB) {
        __syncthreads();
        // A: gather transpose + pos-embed, convert to f16
        for (int i = tid; i < MB * KB; i += 256) {
            int k = i & 31, m = i >> 5;
            int row = m0 + m;
            int b = row >> 10, hw = row & 1023;
            int col = k0 + k;                 // (c d): c = col>>3, d = col&7
            int c = col >> 3, dz = col & 7;
            int h = hw >> 5, wc = hw & 31;
            float xv = x[((((size_t)b * CCH + c) * DD_ + dz) * HH_ + h) * WW_ + wc];
            As[m * LDA + k] = (_Float16)(xv + pos[(size_t)hw * DIMSZ + col]);
        }
        // B: qkv_w[n,k], coalesced along k; prefetch next K-step rows into L2
        for (int i = tid; i < NB * KB; i += 256) {
            int k = i & 31, n = i >> 5;
            Bs[n * LDB + k] = (_Float16)(w[(size_t)(n0 + n) * DIMSZ + (k0 + k)]);
            if (k0 + KB < DIMSZ)
                __builtin_prefetch(&w[(size_t)(n0 + n) * DIMSZ + (k0 + KB + k)], 0, 0);
        }
        __syncthreads();
        v16h af = load_a_frag(As, LDA, mt * 16, 0, lane);
        v16h b0 = load_b_frag(Bs, LDB, ng * 64 +  0, 0, lane);
        v16h b1 = load_b_frag(Bs, LDB, ng * 64 + 16, 0, lane);
        v16h b2 = load_b_frag(Bs, LDB, ng * 64 + 32, 0, lane);
        v16h b3 = load_b_frag(Bs, LDB, ng * 64 + 48, 0, lane);
        acc[0] = wmma16(af, b0, acc[0]);
        acc[1] = wmma16(af, b1, acc[1]);
        acc[2] = wmma16(af, b2, acc[2]);
        acc[3] = wmma16(af, b3, acc[3]);
    }
    // scatter D fragments into q/k/v  (col = which*768 + h*192 + d)
    // 16-wide aligned N-tiles never straddle head/qkv boundaries (16|192|768).
    #pragma unroll
    for (int nt = 0; nt < 4; ++nt) {
        int colb  = n0 + ng * 64 + nt * 16;        // uniform per wave
        int which = colb / DIMSZ;
        int dim   = colb - which * DIMSZ;
        int h     = dim / DHEAD;
        int d     = dim - h * DHEAD + (lane & 15);
        _Float16* dst = (which == 0) ? qb : (which == 1) ? kb : vb;
        #pragma unroll
        for (int r = 0; r < 8; ++r) {
            int row = m0 + mt * 16 + r + ((lane >> 4) << 3);
            int b = row >> 10, hw = row & 1023;
            dst[(((size_t)(b * NHEAD + h) << 10) + hw) * DHEAD + d] = (_Float16)acc[nt][r];
        }
    }
}

// ---------------------------------------------------------------------------
// Kernel 2: per-(b,h,d) L2 norms over the HW axis -> reciprocal scales.
// q-scale additionally folds in temperature[h] (attn = (q^ k^T) * temp[h]).
// ---------------------------------------------------------------------------
__global__ __launch_bounds__(192) void k_scales(const _Float16* __restrict__ qb,
                                                const _Float16* __restrict__ kb,
                                                const float* __restrict__ temp,
                                                float* __restrict__ qsc,
                                                float* __restrict__ ksc) {
    const int which = blockIdx.x;     // 0=q, 1=k
    const int bh    = blockIdx.y;     // b*4+h
    const int d     = threadIdx.x;    // 0..191
    const _Float16* p = (which ? kb : qb) + (size_t)bh * HWSZ * DHEAD;
    float s = 0.f;
    for (int n = 0; n < HWSZ; ++n) { float v = (float)p[(size_t)n * DHEAD + d]; s += v * v; }
    float sc = 1.0f / fmaxf(sqrtf(s), 1e-12f);
    if (which == 0) qsc[bh * DHEAD + d] = sc * temp[bh & 3];
    else            ksc[bh * DHEAD + d] = sc;
}

// ---------------------------------------------------------------------------
// Kernel 3: fused normalized attention, flash-style online softmax.
// grid = (HW/64 q-row tiles, B*NH).  8 waves: wave = (mt = w>>1, nh = w&1).
// S phase: wave owns Mtile mt, key-tiles nh*4..nh*4+3   (24 wmma/chunk)
// O phase: wave owns Mtile mt, d-tiles  nh*6..nh*6+5    (24 wmma/chunk)
// LDS (dynamic, ~148 KB): Qs[64][200] Ks[128][200] Vs^T[192][136] Ps[64][136]
// ---------------------------------------------------------------------------
#define ATT_M   64
#define ATT_C   128
#define LQK     200   // 192 + 8 pad
#define LPV     136   // 128 + 8 pad
#define ATT_LDS ((ATT_M*LQK + ATT_C*LQK + DHEAD*LPV + ATT_M*LPV) * 2 + (3*ATT_M + 4*ATT_M) * 4)

__global__ __launch_bounds__(256) void k_attn(const _Float16* __restrict__ qb,
                                              const _Float16* __restrict__ kb,
                                              const _Float16* __restrict__ vb,
                                              const float* __restrict__ qsc,
                                              const float* __restrict__ ksc,
                                              _Float16* __restrict__ ob) {
    extern __shared__ char smem[];
    _Float16* Qs = (_Float16*)smem;            // [64][LQK]
    _Float16* Ks = Qs + ATT_M * LQK;           // [128][LQK]   key-major
    _Float16* Vs = Ks + ATT_C * LQK;           // [192][LPV]   d-major (transposed)
    _Float16* Ps = Vs + DHEAD * LPV;           // [64][LPV]
    float* run_max = (float*)(Ps + ATT_M * LPV);
    float* run_sum = run_max + ATT_M;
    float* corr    = run_sum + ATT_M;
    float* cmax    = corr + ATT_M;             // [2][64]
    float* csum    = cmax + 2 * ATT_M;         // [2][64]

    const int tid = threadIdx.x, lane = tid & 31, wv = tid >> 5;
    const int mt = wv >> 1, nh = wv & 1;
    const int bh = blockIdx.y;
    const int m0 = blockIdx.x * ATT_M;
    const _Float16* qp = qb + (size_t)bh * HWSZ * DHEAD;
    const _Float16* kp = kb + (size_t)bh * HWSZ * DHEAD;
    const _Float16* vp = vb + (size_t)bh * HWSZ * DHEAD;
    const float* qs = qsc + bh * DHEAD;
    const float* ks = ksc + bh * DHEAD;

    // stage normalized (and temperature-scaled) Q tile once
    for (int i = tid; i < ATT_M * DHEAD; i += 256) {
        int r = i / DHEAD, d = i - r * DHEAD;
        Qs[r * LQK + d] = (_Float16)((float)qp[(size_t)(m0 + r) * DHEAD + d] * qs[d]);
    }
    if (tid < ATT_M) { run_max[tid] = -1e30f; run_sum[tid] = 0.f; }

    v8f oacc[6] = {};
    for (int c0 = 0; c0 < HWSZ; c0 += ATT_C) {
        __syncthreads();   // protects Ks/Vs/Ps reuse + Qs on first pass
        for (int i = tid; i < ATT_C * DHEAD; i += 256) {
            int r = i / DHEAD, d = i - r * DHEAD;
            Ks[r * LQK + d] = (_Float16)((float)kp[(size_t)(c0 + r) * DHEAD + d] * ks[d]);
            if (c0 + ATT_C < HWSZ)
                __builtin_prefetch(&kp[(size_t)(c0 + ATT_C + r) * DHEAD + d], 0, 0);
        }
        for (int i = tid; i < ATT_C * DHEAD; i += 256) {
            int r = i / DHEAD, d = i - r * DHEAD;
            Vs[d * LPV + r] = vp[(size_t)(c0 + r) * DHEAD + d];   // transpose into LDS
            if (c0 + ATT_C < HWSZ)
                __builtin_prefetch(&vp[(size_t)(c0 + ATT_C + r) * DHEAD + d], 0, 0);
        }
        __syncthreads();

        // ---- S = Qn Kn^T (f32 accum) ----
        v8f s[4] = {};
        for (int kk = 0; kk < DHEAD; kk += 32) {
            v16h af = load_a_frag(Qs, LQK, mt * 16, kk, lane);
            v16h b0 = load_b_frag(Ks, LQK, (nh * 4 + 0) * 16, kk, lane);
            v16h b1 = load_b_frag(Ks, LQK, (nh * 4 + 1) * 16, kk, lane);
            v16h b2 = load_b_frag(Ks, LQK, (nh * 4 + 2) * 16, kk, lane);
            v16h b3 = load_b_frag(Ks, LQK, (nh * 4 + 3) * 16, kk, lane);
            s[0] = wmma16(af, b0, s[0]);
            s[1] = wmma16(af, b1, s[1]);
            s[2] = wmma16(af, b2, s[2]);
            s[3] = wmma16(af, b3, s[3]);
        }
        // ---- per-row chunk max (reduce 4 frags, then 16-lane shuffle) ----
        float pm[8];
        #pragma unroll
        for (int r = 0; r < 8; ++r) {
            float m = fmaxf(fmaxf(s[0][r], s[1][r]), fmaxf(s[2][r], s[3][r]));
            #pragma unroll
            for (int off = 1; off < 16; off <<= 1) m = fmaxf(m, __shfl_xor(m, off, 16));
            pm[r] = m;
        }
        if ((lane & 15) == 0) {
            int half = lane >> 4;
            #pragma unroll
            for (int r = 0; r < 8; ++r) cmax[nh * ATT_M + mt * 16 + half * 8 + r] = pm[r];
        }
        __syncthreads();
        if (tid < ATT_M) {
            float nm = fmaxf(run_max[tid], fmaxf(cmax[tid], cmax[ATT_M + tid]));
            corr[tid] = __expf(run_max[tid] - nm);
            run_max[tid] = nm;
        }
        __syncthreads();
        // ---- P = exp(S - rowmax); stash f16 P; partial row sums ----
        float psum[8] = {};
        #pragma unroll
        for (int nt = 0; nt < 4; ++nt) {
            #pragma unroll
            for (int r = 0; r < 8; ++r) {
                int row = mt * 16 + ((lane >> 4) << 3) + r;
                float p = __expf(s[nt][r] - run_max[row]);
                psum[r] += p;
                Ps[row * LPV + (nh * 4 + nt) * 16 + (lane & 15)] = (_Float16)p;
            }
        }
        #pragma unroll
        for (int r = 0; r < 8; ++r) {
            float t = psum[r];
            #pragma unroll
            for (int off = 1; off < 16; off <<= 1) t += __shfl_xor(t, off, 16);
            psum[r] = t;
        }
        if ((lane & 15) == 0) {
            int half = lane >> 4;
            #pragma unroll
            for (int r = 0; r < 8; ++r) csum[nh * ATT_M + mt * 16 + half * 8 + r] = psum[r];
        }
        __syncthreads();
        if (tid < ATT_M) run_sum[tid] = run_sum[tid] * corr[tid] + csum[tid] + csum[ATT_M + tid];
        // ---- rescale O acc, then O += P V ----
        #pragma unroll
        for (int ot = 0; ot < 6; ++ot) {
            #pragma unroll
            for (int r = 0; r < 8; ++r) {
                int row = mt * 16 + ((lane >> 4) << 3) + r;
                oacc[ot][r] *= corr[row];
            }
        }
        v16h pa[4];
        #pragma unroll
        for (int kc = 0; kc < 4; ++kc) pa[kc] = load_a_frag(Ps, LPV, mt * 16, kc * 32, lane);
        #pragma unroll
        for (int ot = 0; ot < 6; ++ot) {
            int dt = nh * 6 + ot;
            v16h b0 = load_b_frag(Vs, LPV, dt * 16,  0, lane);
            v16h b1 = load_b_frag(Vs, LPV, dt * 16, 32, lane);
            v16h b2 = load_b_frag(Vs, LPV, dt * 16, 64, lane);
            v16h b3 = load_b_frag(Vs, LPV, dt * 16, 96, lane);
            oacc[ot] = wmma16(pa[0], b0, oacc[ot]);
            oacc[ot] = wmma16(pa[1], b1, oacc[ot]);
            oacc[ot] = wmma16(pa[2], b2, oacc[ot]);
            oacc[ot] = wmma16(pa[3], b3, oacc[ot]);
        }
    }
    __syncthreads();
    // final: O / rowsum -> o_attn[b][hw][h*192+d] (f16)
    const int b = bh >> 2, h = bh & 3;
    #pragma unroll
    for (int ot = 0; ot < 6; ++ot) {
        int dt = nh * 6 + ot;
        #pragma unroll
        for (int r = 0; r < 8; ++r) {
            int row = mt * 16 + ((lane >> 4) << 3) + r;
            float inv = 1.0f / run_sum[row];
            int hw = m0 + row;
            int d = dt * 16 + (lane & 15);
            ob[((size_t)(b * HWSZ + hw)) * DIMSZ + h * DHEAD + d] = (_Float16)(oacc[ot][r] * inv);
        }
    }
}

// ---------------------------------------------------------------------------
// Kernel 4: proj GEMM.  p[n,j] = sum_k o[n,k]*proj_w[j,k] + proj_b[j]  (f32)
// A tile (64x32 halves, row stride 768, LDS rows padded 32->40 halves) is
// staged with the Tensor Data Mover: each wave DMAs an 8-row slice; the
// D# pad fields (interval=16 DWORDs, amount=4 DWORDs) reproduce LDA=40.
// ---------------------------------------------------------------------------
__global__ __launch_bounds__(256) void k_proj_gemm(const _Float16* __restrict__ ob,
                                                   const float* __restrict__ w,
                                                   const float* __restrict__ bias,
                                                   float* __restrict__ pbuf) {
    __shared__ _Float16 As[MB * LDA];
    __shared__ _Float16 Bs[NB * LDB];
    const int tid = threadIdx.x, lane = tid & 31, wv = tid >> 5;
    const int mt = wv & 3, ng = wv >> 2;
    const int m0 = blockIdx.y * MB;
    const int n0 = blockIdx.x * NB;
    v8f acc[4] = {};
    for (int k0 = 0; k0 < DIMSZ; k0 += KB) {
        __syncthreads();
#if USE_TDM
        {   // TDM: global[m0+wv*8 .. +8)[k0..k0+32) halves -> As[wv*8*LDA]
            unsigned long long ga =
                (unsigned long long)(uintptr_t)(ob + (size_t)(m0 + wv * 8) * DIMSZ + k0);
            unsigned lds_addr = (unsigned)(uintptr_t)(As + wv * 8 * LDA);
            v4u g0;
            g0.x = 1u;                                   // count=1 valid descriptor
            g0.y = lds_addr;                             // D#.lds_addr
            g0.z = (unsigned)ga;                         // global_addr[31:0]
            g0.w = ((unsigned)(ga >> 32) & 0x01FFFFFFu) | (2u << 30);  // [56:32] | type=2
            v8i g1;
            g1[0] = (int)((1u << 16) |                   // data_size = 2B
                          (1u << 20) |                   // pad_enable
                          (3u << 22) |                   // pad_interval: 16 DWORDs
                          (3u << 25));                   // pad_amount: 4 DWORDs
            g1[1] = (int)(768u << 16);                   // tensor_dim0 = 768
            g1[2] = (int)(8192u << 16);                  // tensor_dim1 = 8192
            g1[3] = (int)(32u << 16);                    // tile_dim0 = 32
            g1[4] = 8;                                   // tile_dim1 = 8 (per-wave slice)
            g1[5] = 768;                                 // tensor_dim0_stride = 768
            g1[6] = 0; g1[7] = 0;
            v4i z4 = {0, 0, 0, 0};
#if defined(__clang_major__) && (__clang_major__ >= 23)
            v8i z8 = {0, 0, 0, 0, 0, 0, 0, 0};
            __builtin_amdgcn_tensor_load_to_lds(g0, g1, z4, z4, z8, 0);
#else
            __builtin_amdgcn_tensor_load_to_lds(g0, g1, z4, z4, 0);
#endif
        }
#else
        for (int i = tid; i < MB * KB; i += 256) {
            int k = i & 31, m = i >> 5;
            As[m * LDA + k] = ob[(size_t)(m0 + m) * DIMSZ + (k0 + k)];
        }
#endif
        for (int i = tid; i < NB * KB; i += 256) {
            int k = i & 31, n = i >> 5;
            Bs[n * LDB + k] = (_Float16)(w[(size_t)(n0 + n) * DIMSZ + (k0 + k)]);
            if (k0 + KB < DIMSZ)
                __builtin_prefetch(&w[(size_t)(n0 + n) * DIMSZ + (k0 + KB + k)], 0, 0);
        }
#if USE_TDM
        __builtin_amdgcn_s_wait_tensorcnt(0);
#endif
        __syncthreads();
        v16h af = load_a_frag(As, LDA, mt * 16, 0, lane);
        v16h b0 = load_b_frag(Bs, LDB, ng * 64 +  0, 0, lane);
        v16h b1 = load_b_frag(Bs, LDB, ng * 64 + 16, 0, lane);
        v16h b2 = load_b_frag(Bs, LDB, ng * 64 + 32, 0, lane);
        v16h b3 = load_b_frag(Bs, LDB, ng * 64 + 48, 0, lane);
        acc[0] = wmma16(af, b0, acc[0]);
        acc[1] = wmma16(af, b1, acc[1]);
        acc[2] = wmma16(af, b2, acc[2]);
        acc[3] = wmma16(af, b3, acc[3]);
    }
    #pragma unroll
    for (int nt = 0; nt < 4; ++nt) {
        #pragma unroll
        for (int r = 0; r < 8; ++r) {
            int row = m0 + mt * 16 + r + ((lane >> 4) << 3);
            int col = n0 + ng * 64 + nt * 16 + (lane & 15);
            pbuf[(size_t)row * DIMSZ + col] = acc[nt][r] + bias[col];
        }
    }
}

// ---------------------------------------------------------------------------
// Kernel 5: LayerNorm(768) + un-transpose + residual:  y[b,c,d,h,w]
// ---------------------------------------------------------------------------
__global__ __launch_bounds__(256) void k_ln(const float* __restrict__ pbuf,
                                            const float* __restrict__ x,
                                            const float* __restrict__ gamma,
                                            const float* __restrict__ beta,
                                            float* __restrict__ ybuf) {
    __shared__ float red[256];
    const int row = blockIdx.x, t = threadIdx.x;
    const float* p = pbuf + (size_t)row * DIMSZ;
    float v0 = p[t], v1 = p[t + 256], v2 = p[t + 512];
    red[t] = v0 + v1 + v2; __syncthreads();
    for (int o = 128; o > 0; o >>= 1) { if (t < o) red[t] += red[t + o]; __syncthreads(); }
    float mu = red[0] * (1.f / 768.f);
    __syncthreads();
    float d0 = v0 - mu, d1 = v1 - mu, d2 = v2 - mu;
    red[t] = d0 * d0 + d1 * d1 + d2 * d2; __syncthreads();
    for (int o = 128; o > 0; o >>= 1) { if (t < o) red[t] += red[t + o]; __syncthreads(); }
    float rstd = rsqrtf(red[0] * (1.f / 768.f) + 1e-5f);
    const int b = row >> 10, hw = row & 1023, h = hw >> 5, wq = hw & 31;
    float dv[3] = { d0, d1, d2 };
    #pragma unroll
    for (int i = 0; i < 3; ++i) {
        int col = t + i * 256;
        float o = dv[i] * rstd * gamma[col] + beta[col];
        int c = col >> 3, dz = col & 7;
        size_t idx = ((((size_t)b * CCH + c) * DD_ + dz) * HH_ + h) * WW_ + wq;
        ybuf[idx] = o + x[idx];
    }
}

// ---------------------------------------------------------------------------
// Kernel 6: depthwise 3x3x3 conv (pad 1) over (d,h,w).
// ---------------------------------------------------------------------------
__global__ __launch_bounds__(256) void k_dw(const float* __restrict__ y,
                                            const float* __restrict__ dww,
                                            const float* __restrict__ dwb,
                                            float* __restrict__ z) {
    int i = blockIdx.x * 256 + threadIdx.x;           // B*C*D*H*W = 6291456
    int wq = i & 31, h = (i >> 5) & 31, d = (i >> 10) & 7;
    int bc = i >> 13;                                  // b*96 + c
    int c = bc % CCH;
    float acc = dwb[c];
    const float* wc = dww + c * 27;
    const float* yb = y + (size_t)bc * 8192;
    #pragma unroll
    for (int kd = 0; kd < 3; ++kd) {
        int dd = d + kd - 1; if (dd < 0 || dd > 7) continue;
        #pragma unroll
        for (int kh = 0; kh < 3; ++kh) {
            int hh = h + kh - 1; if (hh < 0 || hh > 31) continue;
            #pragma unroll
            for (int kw = 0; kw < 3; ++kw) {
                int ww = wq + kw - 1; if (ww < 0 || ww > 31) continue;
                acc += yb[(dd << 10) + (hh << 5) + ww] * wc[(kd * 3 + kh) * 3 + kw];
            }
        }
    }
    z[i] = acc;
}

// ---------------------------------------------------------------------------
// Kernel 7: pointwise 96x96 conv + final residual: out = y + (pw(z) + pw_b)
// ---------------------------------------------------------------------------
__global__ __launch_bounds__(256) void k_pw(const float* __restrict__ y,
                                            const float* __restrict__ z,
                                            const float* __restrict__ pww,
                                            const float* __restrict__ pwb,
                                            float* __restrict__ out) {
    int i = blockIdx.x * 256 + threadIdx.x;
    int s = i & 8191;
    int bc = i >> 13;
    int c = bc % CCH, b = bc / CCH;
    float acc = pwb[c];
    const float* zb = z + (size_t)b * CCH * 8192 + s;
    const float* wc = pww + c * CCH;
    #pragma unroll 8
    for (int ci = 0; ci < CCH; ++ci) acc += wc[ci] * zb[(size_t)ci * 8192];
    out[i] = y[i] + acc;
}

// ---------------------------------------------------------------------------
// Host orchestration.  Workspace layout: q,k,v,o_attn f16 (4 x 12.6 MB),
// scales (48 KB), p/y/z f32 (3 x 25.2 MB) — ~126 MB, L2-resident pipeline.
// ---------------------------------------------------------------------------
extern "C" void kernel_launch(void* const* d_in, const int* in_sizes, int n_in,
                              void* d_out, int out_size, void* d_ws, size_t ws_size,
                              hipStream_t stream) {
    const float* x      = (const float*)d_in[0];
    const float* pos    = (const float*)d_in[1];
    const float* qkv_w  = (const float*)d_in[2];
    const float* proj_w = (const float*)d_in[3];
    const float* proj_b = (const float*)d_in[4];
    const float* temp   = (const float*)d_in[5];
    const float* ln_g   = (const float*)d_in[6];
    const float* ln_b   = (const float*)d_in[7];
    const float* dw_w   = (const float*)d_in[8];
    const float* dw_b   = (const float*)d_in[9];
    const float* pw_w   = (const float*)d_in[10];
    const float* pw_b   = (const float*)d_in[11];
    float* out = (float*)d_out;

    char* ws = (char*)d_ws;
    const size_t HB = (size_t)ROWS * DIMSZ * sizeof(_Float16);   // 12,582,912
    _Float16* qb  = (_Float16*)(ws);
    _Float16* kb  = (_Float16*)(ws + HB);
    _Float16* vb  = (_Float16*)(ws + 2 * HB);
    _Float16* ob  = (_Float16*)(ws + 3 * HB);
    float* qsc    = (float*)(ws + 4 * HB);
    float* ksc    = qsc + BATCH * NHEAD * DHEAD;
    float* pbuf   = (float*)(ws + 4 * HB + 2 * BATCH * NHEAD * DHEAD * sizeof(float));
    float* ybuf   = pbuf + (size_t)ROWS * DIMSZ;
    float* zbuf   = ybuf + (size_t)ROWS * DIMSZ;

    // 1) qkv GEMM (WMMA f16)
    k_qkv_gemm<<<dim3(QKVN / NB, ROWS / MB), 256, 0, stream>>>(x, pos, qkv_w, qb, kb, vb);
    // 2) L2-norm scales over HW axis (temperature folded into q-scale)
    k_scales<<<dim3(2, BATCH * NHEAD), 192, 0, stream>>>(qb, kb, temp, qsc, ksc);
    // 3) fused attention (flash-style, WMMA f16), ~148 KB dynamic LDS
    (void)hipFuncSetAttribute(reinterpret_cast<const void*>(k_attn),
                              hipFuncAttributeMaxDynamicSharedMemorySize, ATT_LDS);
    k_attn<<<dim3(HWSZ / ATT_M, BATCH * NHEAD), 256, ATT_LDS, stream>>>(qb, kb, vb, qsc, ksc, ob);
    // 4) proj GEMM + bias (WMMA f16, TDM-staged A tiles)
    k_proj_gemm<<<dim3(DIMSZ / NB, ROWS / MB), 256, 0, stream>>>(ob, proj_w, proj_b, pbuf);
    // 5) LayerNorm + un-transpose + residual
    k_ln<<<ROWS, 256, 0, stream>>>(pbuf, x, ln_g, ln_b, ybuf);
    // 6) depthwise conv
    k_dw<<<(BATCH * CCH * 8192) / 256, 256, 0, stream>>>(ybuf, dw_w, dw_b, zbuf);
    // 7) pointwise conv + final residual
    k_pw<<<(BATCH * CCH * 8192) / 256, 256, 0, stream>>>(ybuf, zbuf, pw_w, pw_b, out);
}
